// VariableDelayGRUMemory_22471268893226
// MI455X (gfx1250) — matrix operand order repaired and checked by reading, not compiled, not run on hardware
//
#include <hip/hip_runtime.h>

// ---------------------------------------------------------------------------
// Variable-delay GRU memory for MI455X (gfx1250, wave32, WMMA, TDM).
//
//   zero_state    : clears hidden state + barrier counters (every launch)
//   pack_whh      : W_hh fp32 -> bf16 WMMA B-fragment layout (192x32 tiles)
//   pack_wdec     : W_dec fp32 -> bf16 B-fragments (2x32 tiles, N padded)
//   gru_persistent: 256 wave32 workgroups (4 batch groups x 64 column slices).
//                   Each wave TDM-loads its 96KB W_hh slice into LDS once,
//                   then runs 1024 steps: 96 v_wmma_f32_16x16x32_bf16/step
//                   with B from LDS (ds_load_b128) and A (hidden state) from
//                   L2; device-scope split barrier per group per step.
//   decode_kernel : delayed gather + (B*T,27) = y_hat @ W_dec^T via WMMA.
// ---------------------------------------------------------------------------

#define BATCH   64
#define TLEN    1024
#define HID     1024
#define NIN     27
#define NGROUP  4       // batch groups of 16 (WMMA M)
#define WGPG    64      // workgroups per group, 16 h-columns each
#define KTILES  32      // 1024 / 32 (WMMA K)

typedef __attribute__((ext_vector_type(16))) __bf16 v16bf;
typedef __attribute__((ext_vector_type(8)))  float  v8f;
typedef __attribute__((ext_vector_type(4)))  unsigned int v4u;
typedef __attribute__((ext_vector_type(8)))  int    v8i_;
typedef __attribute__((ext_vector_type(4)))  int    v4i_;

__device__ __forceinline__ unsigned short f2bf(float f) {
  union { float f; unsigned u; } v; v.f = f;
  unsigned r = v.u + 0x7FFFu + ((v.u >> 16) & 1u);   // round-to-nearest-even
  return (unsigned short)(r >> 16);
}

union AFrag { uint4 q[2]; v16bf v; };

// ---------------------------------------------------------------------------
// TDM: DMA a 2D tile (3 rows x 16384 bf16, row stride 1048576 elems) from
// global fragment buffer into LDS. D# layout per CDNA5 ISA ch.8.
// ---------------------------------------------------------------------------
__device__ __forceinline__ void tdm_load_whh(unsigned lds_byte_addr,
                                             const unsigned short* gsrc) {
#if __has_builtin(__builtin_amdgcn_tensor_load_to_lds)
  unsigned long long ga = (unsigned long long)(size_t)gsrc;
  v4u g0;
  g0.x = 0x1u;                                         // count=1, user mode
  g0.y = lds_byte_addr;                                // lds_addr
  g0.z = (unsigned)(ga & 0xFFFFFFFFu);                 // global_addr[31:0]
  g0.w = (unsigned)((ga >> 32) & 0x01FFFFFFu) | (2u << 30);  // [56:32], type=2
  v8i_ g1;
  g1[0] = 0x10000;              // workgroup_mask=0, data_size=1 (2 bytes)
  g1[1] = (int)(16384u << 16);  // tensor_dim0[15:0]=16384 (bits 63:48)
  g1[2] = (int)(3u << 16);      // tensor_dim0 hi=0 | tensor_dim1[15:0]=3
  g1[3] = (int)(16384u << 16);  // tensor_dim1 hi=0 | tile_dim0=16384
  g1[4] = 3;                    // tile_dim1=3, tile_dim2=0
  g1[5] = 0x100000;             // tensor_dim0_stride = 1048576 elems
  g1[6] = 0;                    // stride0 hi / stride1 lo
  g1[7] = 0;
  v4i_ gz = {0, 0, 0, 0};
#if __clang_major__ >= 23
  v8i_ gz8 = {0, 0, 0, 0, 0, 0, 0, 0};
  __builtin_amdgcn_tensor_load_to_lds(g0, g1, gz, gz, gz8, 0);
#else
  __builtin_amdgcn_tensor_load_to_lds(g0, g1, gz, gz, 0);
#endif
#else
  // Fallback: plain copy (no TDM builtin on this toolchain)
  unsigned short* dst = (unsigned short*)(size_t)lds_byte_addr;
  for (int r = 0; r < 3; ++r)
    for (int j = (int)threadIdx.x; j < 16384; j += 32)
      dst[r * 16384 + j] = gsrc[(size_t)r * 1048576 + j];
#endif
}

// ---------------------------------------------------------------------------
// Prep kernels
// ---------------------------------------------------------------------------
__global__ void zero_state(unsigned short* __restrict__ hbuf,
                           int* __restrict__ counters) {
  int idx = blockIdx.x * blockDim.x + threadIdx.x;
  if (idx < NGROUP * 16 * HID) hbuf[idx] = 0;
  if (idx < NGROUP) counters[idx] = 0;
}

// dst element layout: ((nt*32 + kt)*32 + lane)*16 + i
//   lane 0-15 : column n = nt*16 + lane, k = kt*32 + (i<8 ? i : 8+i)      (i = 0..15)
//   lane 16-31: column n = nt*16 + lane-16, k = kt*32 + 8 + (i<8 ? i : 8+i)
__global__ void pack_whh(const float* __restrict__ W, unsigned short* __restrict__ dst) {
  size_t idx = (size_t)blockIdx.x * blockDim.x + threadIdx.x;
  if (idx >= (size_t)3 * HID * HID) return;
  int i    = (int)(idx & 15);
  int lane = (int)((idx >> 4) & 31);
  int kt   = (int)((idx >> 9) & 31);
  int nt   = (int)(idx >> 14);                  // 0..191
  int n = nt * 16 + (lane & 15);
  int k = kt * 32 + (lane >> 4) * 8 + (i < 8 ? i : 8 + i);
  dst[idx] = f2bf(W[(size_t)n * HID + k]);
}

__global__ void pack_wdec(const float* __restrict__ W, unsigned short* __restrict__ dst) {
  int idx = blockIdx.x * blockDim.x + threadIdx.x;
  if (idx >= 2 * 32 * 512) return;
  int i    = idx & 15;
  int lane = (idx >> 4) & 31;
  int kt   = (idx >> 9) & 31;
  int nt   = idx >> 14;                         // 0..1
  int o = nt * 16 + (lane & 15);
  int k = kt * 32 + (lane >> 4) * 8 + (i < 8 ? i : 8 + i);
  dst[idx] = (o < NIN) ? f2bf(W[(size_t)o * HID + k]) : (unsigned short)0;
}

// ---------------------------------------------------------------------------
// Persistent GRU recurrence. One wave32 per block.
// block = bg*64 + hs : batch group bg (16 rows), h-columns [hs*16, hs*16+16).
// ---------------------------------------------------------------------------
__global__ __launch_bounds__(32)
void gru_persistent(const float* __restrict__ x,
                    const float* __restrict__ W_ih,
                    const float* __restrict__ b_ih,
                    const float* __restrict__ b_hh,
                    const unsigned short* __restrict__ whh_frag,
                    unsigned short* __restrict__ hbuf,   // [4][16][1024] bf16
                    unsigned short* __restrict__ ybuf,   // [64][1024][1024] bf16
                    int* __restrict__ counters) {        // [4]
  const int tid = threadIdx.x;
  const int bg  = blockIdx.x / WGPG;
  const int hs  = blockIdx.x % WGPG;
  const int cc  = tid & 15;       // column-in-slice (C layout N), A row index
  const int hi  = tid >> 4;       // half select
  const int mo  = hi * 8;         // C-layout row offset: m = mo + j

  __shared__ unsigned short sWhh[3 * KTILES * 32 * 16];  // 96 KB: B fragments
  __shared__ float sW[3 * 16 * NIN];   // W_ih slice for our 48 gate columns
  __shared__ float sx[16 * NIN];       // x[batch tile, t, :]

  // One-shot DMA of this WG's W_hh fragment slice into LDS (3 x 32KB rows).
  tdm_load_whh((unsigned)(size_t)&sWhh[0], whh_frag + (size_t)hs * 16384);

  for (int idx = tid; idx < 3 * 16 * NIN; idx += 32) {
    int g = idx / (16 * NIN);
    int r = idx - g * 16 * NIN;
    int c = r / NIN;
    int i = r - c * NIN;
    sW[idx] = W_ih[(size_t)(g * HID + hs * 16 + c) * NIN + i];
  }

  const int col_r = 0 * HID + hs * 16 + cc;
  const int col_z = 1 * HID + hs * 16 + cc;
  const int col_n = 2 * HID + hs * 16 + cc;
  const float bias_r  = b_ih[col_r] + b_hh[col_r];
  const float bias_z  = b_ih[col_z] + b_hh[col_z];
  const float bias_in = b_ih[col_n];      // i_n part (outside r*h_n)
  const float bias_hn = b_hh[col_n];      // h_n part (multiplied by r)

  v8f hprev = {};                         // our slice of h (C layout), h0 = 0

  const unsigned* hrows = (const unsigned*)hbuf;             // dword view
  const size_t hdw = (size_t)bg * 16 * (HID / 2);            // group base
  const size_t arow = hdw + (size_t)cc * (HID / 2) + (hi ? 4 : 0);
  int* cnt = counters + bg;

  __builtin_amdgcn_s_wait_tensorcnt(0);   // TDM complete before first B read
  __syncthreads();

  for (int t = 0; t < TLEN; ++t) {
    // ---- stage x tile --------------------------------------------------
    for (int idx = tid; idx < 16 * NIN; idx += 32) {
      int m = idx / NIN, i = idx - m * NIN;
      sx[idx] = x[((size_t)(bg * 16 + m) * TLEN + t) * NIN + i];
    }
    // prefetch next step's x rows while we compute this one
    if (t + 1 < TLEN && tid < 16)
      __builtin_prefetch(&x[((size_t)(bg * 16 + tid) * TLEN + (t + 1)) * NIN], 0, 3);
    __syncthreads();

    // ---- gi = b + W_ih @ x_t  (K=27, VALU) -----------------------------
    v8f cr, cz, cn, gin;
    #pragma unroll
    for (int j = 0; j < 8; ++j) { cr[j] = bias_r; cz[j] = bias_z; gin[j] = bias_in; cn[j] = bias_hn; }
    #pragma unroll 1
    for (int i = 0; i < NIN; ++i) {
      float wr = sW[(0 * 16 + cc) * NIN + i];
      float wz = sW[(1 * 16 + cc) * NIN + i];
      float wn = sW[(2 * 16 + cc) * NIN + i];
      #pragma unroll
      for (int j = 0; j < 8; ++j) {
        float xv = sx[(mo + j) * NIN + i];
        cr[j]  = __fmaf_rn(wr, xv, cr[j]);
        cz[j]  = __fmaf_rn(wz, xv, cz[j]);
        gin[j] = __fmaf_rn(wn, xv, gin[j]);
      }
    }

    // ---- gh += h @ W_hh^T over K=1024 via WMMA bf16, B from LDS --------
    #pragma unroll 4
    for (int kt = 0; kt < KTILES; ++kt) {
      AFrag a;                                   // A: 16x32 slice of h
      const unsigned* p = hrows + arow + kt * 16;
      a.q[0] = *(const uint4*)(p);               // k 0..7  (this half)
      a.q[1] = *(const uint4*)(p + 8);           // k 16..23 (this half)
      const int lb = ((kt * 32) + tid) * 16;
      v16bf br = *(const v16bf*)(sWhh + 0 * (KTILES * 512) + lb);
      v16bf bz = *(const v16bf*)(sWhh + 1 * (KTILES * 512) + lb);
      v16bf bn = *(const v16bf*)(sWhh + 2 * (KTILES * 512) + lb);
      cr = __builtin_amdgcn_wmma_f32_16x16x32_bf16(false, a.v, false, br, (short)0, cr, false, false);
      cz = __builtin_amdgcn_wmma_f32_16x16x32_bf16(false, a.v, false, bz, (short)0, cz, false, false);
      cn = __builtin_amdgcn_wmma_f32_16x16x32_bf16(false, a.v, false, bn, (short)0, cn, false, false);
    }

    // ---- gate math (elementwise in C layout) ---------------------------
    v8f hnew;
    #pragma unroll
    for (int j = 0; j < 8; ++j) {
      float r = 1.0f / (1.0f + __expf(-cr[j]));
      float z = 1.0f / (1.0f + __expf(-cz[j]));
      float npre = gin[j] + r * cn[j];
      float e = __expf(2.0f * npre);
      float nact = (e - 1.0f) / (e + 1.0f);
      hnew[j] = (1.0f - z) * nact + z * hprev[j];
    }
    hprev = hnew;

    // ---- publish h slice (bf16) + y_hat --------------------------------
    const int ncol = hs * 16 + cc;
    #pragma unroll
    for (int j = 0; j < 8; ++j) {
      int m = mo + j;
      unsigned short hb = f2bf(hnew[j]);
      hbuf[(size_t)(bg * 16 + m) * HID + ncol] = hb;
      ybuf[((size_t)(bg * 16 + m) * TLEN + t) * HID + ncol] = hb;
    }

    // ---- device-scope split barrier across the group's 64 waves --------
    __threadfence();
    __syncthreads();
    if (tid == 0) {
      __hip_atomic_fetch_add(cnt, 1, __ATOMIC_RELEASE, __HIP_MEMORY_SCOPE_AGENT);
      const int target = WGPG * (t + 1);
      while (__hip_atomic_load(cnt, __ATOMIC_ACQUIRE, __HIP_MEMORY_SCOPE_AGENT) < target)
        __builtin_amdgcn_s_sleep(1);
    }
    __syncthreads();
  }
}

// ---------------------------------------------------------------------------
// Decode: logits[b,t,:] = y_hat[b, delay(t), :] @ W_dec^T + b_dec   (K=1024)
// One wave32 per (b, 16-row t-tile); delayed row gather folded into A rows.
// ---------------------------------------------------------------------------
__global__ __launch_bounds__(32)
void decode_kernel(const unsigned short* __restrict__ ybuf,
                   const unsigned short* __restrict__ wdec_frag,
                   const float* __restrict__ b_dec,
                   const int* __restrict__ delays,
                   float* __restrict__ out) {
  const int tid = threadIdx.x;
  const int b  = blockIdx.x >> 6;
  const int t0 = (blockIdx.x & 63) * 16;
  const int cc = tid & 15;
  const int hi = tid >> 4;
  const int mo = hi * 8;

  const int d = delays[b];
  const int trow = t0 + cc;                       // A row this lane feeds
  const int td = (trow < d) ? trow : trow - d;    // delayed index
  const unsigned* yd = (const unsigned*)ybuf;
  const size_t arow = ((size_t)b * TLEN + td) * (HID / 2) + (hi ? 4 : 0);

  const int o1 = 16 + cc;
  const float bias0 = b_dec[cc];
  const float bias1 = (o1 < NIN) ? b_dec[o1] : 0.0f;
  v8f c0, c1;
  #pragma unroll
  for (int j = 0; j < 8; ++j) { c0[j] = bias0; c1[j] = bias1; }

  #pragma unroll 4
  for (int kt = 0; kt < KTILES; ++kt) {
    AFrag a;
    const unsigned* p = yd + arow + kt * 16;
    a.q[0] = *(const uint4*)(p);
    a.q[1] = *(const uint4*)(p + 8);
    const size_t lb = (size_t)tid * 16;
    v16bf b0 = *(const v16bf*)(wdec_frag + (((size_t)(0 * 32 + kt) * 32) * 16) + lb);
    v16bf b1 = *(const v16bf*)(wdec_frag + (((size_t)(1 * 32 + kt) * 32) * 16) + lb);
    c0 = __builtin_amdgcn_wmma_f32_16x16x32_bf16(false, a.v, false, b0, (short)0, c0, false, false);
    c1 = __builtin_amdgcn_wmma_f32_16x16x32_bf16(false, a.v, false, b1, (short)0, c1, false, false);
  }

  #pragma unroll
  for (int j = 0; j < 8; ++j) {
    int m = mo + j;
    size_t base = ((size_t)b * TLEN + (t0 + m)) * NIN;
    out[base + cc] = c0[j];
    if (o1 < NIN) out[base + o1] = c1[j];
  }
}

// ---------------------------------------------------------------------------
// Launch. Workspace layout (bytes, 256B-aligned sections, ~140.7 MB total):
//   [0)            W_hh bf16 fragments   6,291,456
//   [6,291,456)    W_dec bf16 fragments     65,536
//   [6,356,992)    hidden state bf16       131,072
//   [6,488,064)    barrier counters            256
//   [6,488,320)    y_hat bf16          134,217,728
// ---------------------------------------------------------------------------
extern "C" void kernel_launch(void* const* d_in, const int* in_sizes, int n_in,
                              void* d_out, int out_size, void* d_ws, size_t ws_size,
                              hipStream_t stream) {
  (void)in_sizes; (void)n_in; (void)out_size; (void)ws_size;
  const float* x      = (const float*)d_in[0];
  const int*   delays = (const int*)d_in[1];
  const float* W_ih   = (const float*)d_in[2];
  const float* W_hh   = (const float*)d_in[3];
  const float* b_ih   = (const float*)d_in[4];
  const float* b_hh   = (const float*)d_in[5];
  const float* W_dec  = (const float*)d_in[6];
  const float* b_dec  = (const float*)d_in[7];
  float* out = (float*)d_out;

  char* ws = (char*)d_ws;
  unsigned short* whh_frag  = (unsigned short*)(ws + 0);
  unsigned short* wdec_frag = (unsigned short*)(ws + 6291456);
  unsigned short* hbuf      = (unsigned short*)(ws + 6356992);
  int*            counters  = (int*)          (ws + 6488064);
  unsigned short* ybuf      = (unsigned short*)(ws + 6488320);

  zero_state<<<dim3(256), dim3(256), 0, stream>>>(hbuf, counters);
  pack_whh<<<dim3((3 * HID * HID + 255) / 256), dim3(256), 0, stream>>>(W_hh, whh_frag);
  pack_wdec<<<dim3((2 * 32 * 512 + 255) / 256), dim3(256), 0, stream>>>(W_dec, wdec_frag);
  gru_persistent<<<dim3(NGROUP * WGPG), dim3(32), 0, stream>>>(
      x, W_ih, b_ih, b_hh, whh_frag, hbuf, ybuf, counters);
  decode_kernel<<<dim3(BATCH * (TLEN / 16)), dim3(32), 0, stream>>>(
      ybuf, wdec_frag, b_dec, delays, out);
}
